// Generator_56702158241891
// MI455X (gfx1250) — matrix operand order repaired and checked by reading, not compiled, not run on hardware
//
#include <hip/hip_runtime.h>
#include <hip/hip_bf16.h>

// ---------------------------------------------------------------------------
// Types for CDNA5 WMMA
// ---------------------------------------------------------------------------
typedef __attribute__((ext_vector_type(16))) __bf16 v16bf;
typedef __attribute__((ext_vector_type(8)))  float  v8f;

__device__ __forceinline__ __bf16 f2bf(float f) {
    // round-to-nearest-even fp32 -> bf16, pure bit manipulation
    unsigned u = __builtin_bit_cast(unsigned, f);
    unsigned r = u + 0x7FFFu + ((u >> 16) & 1u);
    unsigned short h = (unsigned short)(r >> 16);
    return __builtin_bit_cast(__bf16, h);
}

// ---------------------------------------------------------------------------
// Weight packing: w[Cout, K] (f32) -> A-fragment layout (bf16)
//   wp[ct][kt][lane][e],  co = ct*16 + (lane&15)
//   k  = kt*32 + e + (e>=8 ? 8:0) + (lane>>4)*8      (16-bit A 16x32 layout)
// ---------------------------------------------------------------------------
__global__ void pack_w_kernel(const float* __restrict__ w, __bf16* __restrict__ wp,
                              int Cout, int Kdim, int nkt, size_t total)
{
    for (size_t idx = (size_t)blockIdx.x * blockDim.x + threadIdx.x; idx < total;
         idx += (size_t)gridDim.x * blockDim.x) {
        int e    = (int)(idx & 15);
        int lane = (int)((idx >> 4) & 31);
        size_t t = idx >> 9;
        int kt   = (int)(t % nkt);
        int ct   = (int)(t / nkt);
        int co   = ct * 16 + (lane & 15);
        int k    = kt * 32 + e + ((e >= 8) ? 8 : 0) + (lane >> 4) * 8;
        float v  = (co < Cout && k < Kdim) ? w[(size_t)co * Kdim + k] : 0.f;
        wp[idx] = f2bf(v);
    }
}

// ---------------------------------------------------------------------------
// im2col packing: x[N,Cin,H,W] (f32) -> B-fragment layout (bf16), pixel-chunked
//   xp[n][ptl][kt][lane][e],  p = (pt0+ptl)*16 + (lane&15)
//   k = kt*32 + e + (e>=8 ? 8:0) + (lane>>4)*8       (16-bit B 32x16 layout)
// ---------------------------------------------------------------------------
__global__ void pack_im2col_kernel(const float* __restrict__ x, __bf16* __restrict__ xp,
                                   int N, int Cin, int H, int W,
                                   int KH, int KW, int stride, int pad,
                                   int Wo, int Npix, int Kdim, int nkt,
                                   int pt0, int nptc, size_t total)
{
    const int KHW = KH * KW;
    for (size_t idx = (size_t)blockIdx.x * blockDim.x + threadIdx.x; idx < total;
         idx += (size_t)gridDim.x * blockDim.x) {
        int e    = (int)(idx & 15);
        int lane = (int)((idx >> 4) & 31);
        size_t t = idx >> 9;
        int kt   = (int)(t % nkt);  t /= nkt;
        int ptl  = (int)(t % nptc);
        int n    = (int)(t / nptc);
        int p    = (pt0 + ptl) * 16 + (lane & 15);
        int k    = kt * 32 + e + ((e >= 8) ? 8 : 0) + (lane >> 4) * 8;
        float v  = 0.f;
        if (p < Npix && k < Kdim) {
            int oh = p / Wo, ow = p - oh * Wo;
            int ci = k / KHW;
            int r  = k - ci * KHW;
            int kh = r / KW, kw = r - kh * KW;
            int ih = oh * stride - pad + kh;
            int iw = ow * stride - pad + kw;
            if (ih >= 0 && ih < H && iw >= 0 && iw < W)
                v = x[(((size_t)n * Cin + ci) * H + ih) * W + iw];
        }
        xp[idx] = f2bf(v);
    }
}

// ---------------------------------------------------------------------------
// D-tile store: vgpr r -> row M = r + 8*(lane>=16); col N = lane&15
// ---------------------------------------------------------------------------
__device__ __forceinline__ void store_tile(const v8f& acc, float* __restrict__ y,
                                           const float* __restrict__ bias,
                                           int n, int ct, int ptAbs,
                                           int Cout, int Npix, int lane)
{
    const int p = ptAbs * 16 + (lane & 15);
    if (p >= Npix) return;
    const int half = lane >> 4;
#pragma unroll
    for (int r = 0; r < 8; ++r) {
        int co = ct * 16 + half * 8 + r;
        if (co < Cout) {
            float v = acc[r];
            if (bias) v += bias[co];
            y[((size_t)n * Cout + co) * Npix + p] = v;
        }
    }
}

// ---------------------------------------------------------------------------
// Register-tiled GEMM on packed fragments: each wave computes a 2x2 grid of
// 16x16 tiles (32 Cout x 32 pixels). Per k-step: 2 A-frags + 2 B-frags
// (8x global_load_b128) feed 4 WMMAs -> 2 loads per WMMA.
// 4 waves/block share the same B-frag pair (different Cout pairs) -> L0 hits.
// Edge tiles are clamped (duplicate, idempotent stores) so EXEC stays uniform.
// ---------------------------------------------------------------------------
__global__ void gemm_wmma_kernel(const __bf16* __restrict__ Ap,
                                 const __bf16* __restrict__ Bp,
                                 const float* __restrict__ bias,
                                 float* __restrict__ y,
                                 int nkt, int nct, int Cout, int Npix,
                                 int pt0, int nptc)
{
    const int lane = threadIdx.x & 31;
    const int wid  = threadIdx.x >> 5;
    const int ct0  = blockIdx.x * 8 + wid * 2;
    if (ct0 >= nct) return;          // whole-wave exit: EXEC all-ones for WMMA
    int ct1 = ct0 + 1; if (ct1 >= nct) ct1 = nct - 1;
    const int pl0 = blockIdx.y * 2;
    int pl1 = pl0 + 1; if (pl1 >= nptc) pl1 = nptc - 1;
    const int n = blockIdx.z;

    const v16bf* a0p = (const v16bf*)Ap + ((size_t)ct0 * nkt) * 32 + lane;
    const v16bf* a1p = (const v16bf*)Ap + ((size_t)ct1 * nkt) * 32 + lane;
    const v16bf* b0p = (const v16bf*)Bp + (((size_t)n * nptc + pl0) * nkt) * 32 + lane;
    const v16bf* b1p = (const v16bf*)Bp + (((size_t)n * nptc + pl1) * nkt) * 32 + lane;

    v8f acc00 = {0.f, 0.f, 0.f, 0.f, 0.f, 0.f, 0.f, 0.f};
    v8f acc01 = acc00, acc10 = acc00, acc11 = acc00;

    for (int kt = 0; kt < nkt; ++kt) {
        v16bf a0 = a0p[(size_t)kt * 32];
        v16bf a1 = a1p[(size_t)kt * 32];
        v16bf b0 = b0p[(size_t)kt * 32];
        v16bf b1 = b1p[(size_t)kt * 32];
        acc00 = __builtin_amdgcn_wmma_f32_16x16x32_bf16(false, a0, false, b0,
                                                        (short)0, acc00, false, false);
        acc01 = __builtin_amdgcn_wmma_f32_16x16x32_bf16(false, a0, false, b1,
                                                        (short)0, acc01, false, false);
        acc10 = __builtin_amdgcn_wmma_f32_16x16x32_bf16(false, a1, false, b0,
                                                        (short)0, acc10, false, false);
        acc11 = __builtin_amdgcn_wmma_f32_16x16x32_bf16(false, a1, false, b1,
                                                        (short)0, acc11, false, false);
    }

    store_tile(acc00, y, bias, n, ct0, pt0 + pl0, Cout, Npix, lane);
    store_tile(acc01, y, bias, n, ct0, pt0 + pl1, Cout, Npix, lane);
    store_tile(acc10, y, bias, n, ct1, pt0 + pl0, Cout, Npix, lane);
    store_tile(acc11, y, bias, n, ct1, pt0 + pl1, Cout, Npix, lane);
}

// ---------------------------------------------------------------------------
// Instance norm (+ optional ReLU), one block per (n, c)
// ---------------------------------------------------------------------------
__global__ void instnorm_kernel(const float* __restrict__ x, float* __restrict__ y,
                                const float* __restrict__ g, const float* __restrict__ b,
                                int C, int HW, int do_relu)
{
    const int nc = blockIdx.x;
    const int c  = nc % C;
    const float* xp = x + (size_t)nc * HW;
    float* yp = y + (size_t)nc * HW;

    float s = 0.f, s2 = 0.f;
    for (int i = threadIdx.x; i < HW; i += blockDim.x) {
        float v = xp[i];
        s += v; s2 += v * v;
    }
    __shared__ float sh[256], sh2[256];
    sh[threadIdx.x] = s; sh2[threadIdx.x] = s2;
    __syncthreads();
    for (int st = 128; st > 0; st >>= 1) {
        if (threadIdx.x < st) {
            sh[threadIdx.x]  += sh[threadIdx.x + st];
            sh2[threadIdx.x] += sh2[threadIdx.x + st];
        }
        __syncthreads();
    }
    float mu  = sh[0] / (float)HW;
    float var = fmaxf(sh2[0] / (float)HW - mu * mu, 0.f);
    float scale = rsqrtf(var + 1e-5f) * g[c];
    float shift = b[c] - mu * scale;
    for (int i = threadIdx.x; i < HW; i += blockDim.x) {
        float v = xp[i] * scale + shift;
        if (do_relu) v = fmaxf(v, 0.f);
        yp[i] = v;
    }
}

// ---------------------------------------------------------------------------
// Per-(n,c) mean and std (ddof=1)
// ---------------------------------------------------------------------------
__global__ void meanstd_kernel(const float* __restrict__ x,
                               float* __restrict__ mean, float* __restrict__ stdv,
                               int HW)
{
    const int nc = blockIdx.x;
    const float* xp = x + (size_t)nc * HW;
    float s = 0.f, s2 = 0.f;
    for (int i = threadIdx.x; i < HW; i += blockDim.x) {
        float v = xp[i];
        s += v; s2 += v * v;
    }
    __shared__ float sh[256], sh2[256];
    sh[threadIdx.x] = s; sh2[threadIdx.x] = s2;
    __syncthreads();
    for (int st = 128; st > 0; st >>= 1) {
        if (threadIdx.x < st) {
            sh[threadIdx.x]  += sh[threadIdx.x + st];
            sh2[threadIdx.x] += sh2[threadIdx.x + st];
        }
        __syncthreads();
    }
    if (threadIdx.x == 0) {
        float mu = sh[0] / (float)HW;
        float ss = fmaxf(sh2[0] - (float)HW * mu * mu, 0.f);
        mean[nc] = mu;
        stdv[nc] = sqrtf(ss / (float)(HW - 1));
    }
}

// ---------------------------------------------------------------------------
// Sequential running-stat scan over batch (single block; tiny data)
// ---------------------------------------------------------------------------
__global__ void scan_kernel(const float* __restrict__ mean, const float* __restrict__ stdv,
                            const float* __restrict__ c_org,
                            const float* __restrict__ rm_in, const float* __restrict__ rs_in,
                            float* __restrict__ rm_out, float* __restrict__ rs_out,
                            int B, int CD, int C)
{
    const int R = 2 * CD;
    for (int idx = threadIdx.x; idx < R * C; idx += blockDim.x) {
        rm_out[idx] = rm_in[idx];
        rs_out[idx] = rs_in[idx];
    }
    __syncthreads();
    for (int n = 0; n < B; ++n) {
        for (int idx = threadIdx.x; idx < R * C; idx += blockDim.x) {
            int j = idx / C;
            int c = idx - j * C;
            float cv = c_org[n * CD + (j < CD ? j : j - CD)];
            bool m = (j < CD) ? (cv != 0.f) : (cv == 0.f);
            if (m) {
                rm_out[idx] = rm_out[idx] * 0.9f + mean[n * C + c] * 0.1f;
                rs_out[idx] = rs_out[idx] * 0.9f + stdv[n * C + c] * 0.1f;
            }
        }
        __syncthreads();
    }
}

// ---------------------------------------------------------------------------
// trg stats: (ct @ rm_f) / cnt per (n, c)
// ---------------------------------------------------------------------------
__global__ void target_kernel(const float* __restrict__ cvec,
                              const float* __restrict__ rm, const float* __restrict__ rs,
                              float* __restrict__ tmean, float* __restrict__ tstd,
                              int B, int CD, int C)
{
    size_t total = (size_t)B * C;
    for (size_t idx = (size_t)blockIdx.x * blockDim.x + threadIdx.x; idx < total;
         idx += (size_t)gridDim.x * blockDim.x) {
        int n = (int)(idx / C);
        int c = (int)(idx - (size_t)n * C);
        float sm = 0.f, ss = 0.f, cnt = 0.f;
        for (int j = 0; j < 2 * CD; ++j) {
            float cv = cvec[n * CD + (j < CD ? j : j - CD)];
            bool m = (j < CD) ? (cv != 0.f) : (cv == 0.f);
            if (m) { cnt += 1.f; sm += rm[j * C + c]; ss += rs[j * C + c]; }
        }
        tmean[idx] = sm / cnt;
        tstd[idx]  = ss / cnt;
    }
}

__global__ void renorm_kernel(float* __restrict__ x,
                              const float* __restrict__ mean, const float* __restrict__ stdv,
                              const float* __restrict__ tmean, const float* __restrict__ tstd,
                              int HW, size_t total)
{
    for (size_t idx = (size_t)blockIdx.x * blockDim.x + threadIdx.x; idx < total;
         idx += (size_t)gridDim.x * blockDim.x) {
        size_t nc = idx / HW;
        x[idx] = (x[idx] - mean[nc]) / stdv[nc] * tstd[nc] + tmean[nc];
    }
}

__global__ void concat_kernel(const float* __restrict__ x2, const float* __restrict__ c,
                              float* __restrict__ xc, int C1, int C2, int HW, size_t total)
{
    int C = C1 + C2;
    for (size_t idx = (size_t)blockIdx.x * blockDim.x + threadIdx.x; idx < total;
         idx += (size_t)gridDim.x * blockDim.x) {
        int p    = (int)(idx % HW);
        size_t r = idx / HW;
        int ch   = (int)(r % C);
        int n    = (int)(r / C);
        xc[idx] = (ch < C1) ? x2[((size_t)n * C1 + ch) * HW + p]
                            : c[n * C2 + (ch - C1)];
    }
}

__global__ void add_kernel(float* __restrict__ x, const float* __restrict__ h, size_t total)
{
    for (size_t idx = (size_t)blockIdx.x * blockDim.x + threadIdx.x; idx < total;
         idx += (size_t)gridDim.x * blockDim.x)
        x[idx] += h[idx];
}

// ---------------------------------------------------------------------------
// Bilinear 2x upsample, half-pixel centers
// ---------------------------------------------------------------------------
__global__ void resize2x_kernel(const float* __restrict__ x, float* __restrict__ y,
                                int H, int W, size_t total)
{
    int H2 = 2 * H, W2 = 2 * W;
    for (size_t idx = (size_t)blockIdx.x * blockDim.x + threadIdx.x; idx < total;
         idx += (size_t)gridDim.x * blockDim.x) {
        int ow   = (int)(idx % W2);
        size_t r = idx / W2;
        int oh   = (int)(r % H2);
        size_t nc = r / H2;
        const float* xp = x + nc * H * W;

        float fy = (oh + 0.5f) * 0.5f - 0.5f;
        float fx = (ow + 0.5f) * 0.5f - 0.5f;
        int y0 = (int)floorf(fy);
        int x0 = (int)floorf(fx);
        float wy = fy - (float)y0;
        float wx = fx - (float)x0;
        int y0c = min(max(y0, 0), H - 1);
        int y1c = min(max(y0 + 1, 0), H - 1);
        int x0c = min(max(x0, 0), W - 1);
        int x1c = min(max(x0 + 1, 0), W - 1);
        float v00 = xp[(size_t)y0c * W + x0c];
        float v01 = xp[(size_t)y0c * W + x1c];
        float v10 = xp[(size_t)y1c * W + x0c];
        float v11 = xp[(size_t)y1c * W + x1c];
        y[idx] = v00 * (1.f - wy) * (1.f - wx) + v01 * (1.f - wy) * wx
               + v10 * wy * (1.f - wx) + v11 * wy * wx;
    }
}

// ---------------------------------------------------------------------------
// Polynomial recombination
// ---------------------------------------------------------------------------
__global__ void poly_kernel(const float* __restrict__ y, const float* __restrict__ im,
                            float* __restrict__ out, int HW, size_t total)
{
    for (size_t idx = (size_t)blockIdx.x * blockDim.x + threadIdx.x; idx < total;
         idx += (size_t)gridDim.x * blockDim.x) {
        int p    = (int)(idx % HW);
        size_t r = idx / HW;
        int ch   = (int)(r % 3);
        int n    = (int)(r / 3);
        const float* yn = y + (size_t)n * 12 * HW;
        float a0 = yn[(0 * 3 + ch) * (size_t)HW + p];
        float a1 = yn[(1 * 3 + ch) * (size_t)HW + p];
        float a2 = yn[(2 * 3 + ch) * (size_t)HW + p];
        float a3 = yn[(3 * 3 + ch) * (size_t)HW + p];
        float d  = fabsf(a0) + fabsf(a1) + fabsf(a2) + fabsf(a3) + 0.01f;
        float v  = im[idx];
        out[idx] = (a0 + v * (a1 + v * (a2 + v * a3))) / d;
    }
}

// ---------------------------------------------------------------------------
// Host-side orchestration
// ---------------------------------------------------------------------------
static inline int ew_blocks(size_t n) {
    size_t b = (n + 255) / 256;
    if (b > (size_t)1048576) b = 1048576;
    return (int)b;
}

// xpack budget: 80M bf16 elements (160 MB)
#define XPACK_CAP ((size_t)80 * 1024 * 1024)

static void conv_packed(const float* x, const float* w, const float* bias, float* y,
                        int N, int Cin, int H, int W, int Cout, int KH, int KW,
                        int stride, int pad, __bf16* wpack, __bf16* xpack,
                        hipStream_t s)
{
    const int Ho   = (H + 2 * pad - KH) / stride + 1;
    const int Wo   = (W + 2 * pad - KW) / stride + 1;
    const int Npix = Ho * Wo;
    const int Kdim = Cin * KH * KW;
    const int nkt  = (Kdim + 31) / 32;
    const int nct  = (Cout + 15) / 16;
    const int npt  = (Npix + 15) / 16;

    // pack weights (A fragments)
    size_t wtot = (size_t)nct * nkt * 512;
    pack_w_kernel<<<ew_blocks(wtot), 256, 0, s>>>(w, wpack, Cout, Kdim, nkt, wtot);

    // pixel-tile chunking so B-panel fits the xpack budget
    size_t perPt = (size_t)N * nkt * 512;
    int maxnpt = (int)(XPACK_CAP / perPt);
    if (maxnpt < 1) maxnpt = 1;

    for (int pt0 = 0; pt0 < npt; pt0 += maxnpt) {
        int nptc = (npt - pt0 < maxnpt) ? (npt - pt0) : maxnpt;
        size_t btot = (size_t)N * nptc * nkt * 512;
        pack_im2col_kernel<<<ew_blocks(btot), 256, 0, s>>>(
            x, xpack, N, Cin, H, W, KH, KW, stride, pad,
            Wo, Npix, Kdim, nkt, pt0, nptc, btot);
        dim3 grid((nct + 7) / 8, (nptc + 1) / 2, N);
        gemm_wmma_kernel<<<grid, 128, 0, s>>>(wpack, xpack, bias, y,
                                              nkt, nct, Cout, Npix, pt0, nptc);
    }
}

static void launch_in(const float* x, float* y, const float* g, const float* b,
                      int N, int C, int HW, int relu, hipStream_t s)
{
    instnorm_kernel<<<N * C, 256, 0, s>>>(x, y, g, b, C, HW, relu);
}

extern "C" void kernel_launch(void* const* d_in, const int* in_sizes, int n_in,
                              void* d_out, int out_size, void* d_ws, size_t ws_size,
                              hipStream_t stream)
{
    (void)in_sizes; (void)n_in; (void)out_size; (void)ws_size;

    const float* im       = (const float*)d_in[0];
    const float* c        = (const float*)d_in[1];
    const float* c_org    = (const float*)d_in[2];
    const float* conv0_w  = (const float*)d_in[3];
    const float* in0_g    = (const float*)d_in[4];
    const float* in0_b    = (const float*)d_in[5];
    const float* down1_w  = (const float*)d_in[6];
    const float* in1_g    = (const float*)d_in[7];
    const float* in1_b    = (const float*)d_in[8];
    const float* down2_w  = (const float*)d_in[9];
    const float* in2_g    = (const float*)d_in[10];
    const float* in2_b    = (const float*)d_in[11];
    const float* rmean    = (const float*)d_in[12];
    const float* rstd     = (const float*)d_in[13];
    const float* res_w1   = (const float*)d_in[14];
    const float* res_g1   = (const float*)d_in[15];
    const float* res_b1   = (const float*)d_in[16];
    const float* res_w2   = (const float*)d_in[17];
    const float* res_g2   = (const float*)d_in[18];
    const float* res_b2   = (const float*)d_in[19];
    const float* up0_w    = (const float*)d_in[20];
    const float* up0_b    = (const float*)d_in[21];
    const float* up0_g    = (const float*)d_in[22];
    const float* up0_bt   = (const float*)d_in[23];
    const float* up1_w    = (const float*)d_in[24];
    const float* up1_b    = (const float*)d_in[25];
    const float* up1_g    = (const float*)d_in[26];
    const float* up1_bt   = (const float*)d_in[27];
    const float* out_w    = (const float*)d_in[28];
    const float* out_b    = (const float*)d_in[29];
    float* out = (float*)d_out;

    const int B = 8, CD = 5, CUR = 261;  // 64*4 + 5

    // ---- workspace layout (floats, 64-float aligned), lifetime-based reuse ----
    float* Wf = (float*)d_ws;
    size_t off = 0;
    auto alloc = [&](size_t nfloats) {
        float* p = Wf + off;
        off += (nfloats + 63) & ~(size_t)63;
        return p;
    };
    float* mean  = alloc((size_t)B * CUR);
    float* stdv  = alloc((size_t)B * CUR);
    float* rmf   = alloc((size_t)2 * CD * CUR);
    float* rsf   = alloc((size_t)2 * CD * CUR);
    float* tmean = alloc((size_t)B * CUR);
    float* tstd  = alloc((size_t)B * CUR);
    float*  bufA  = alloc((size_t)8 * 261 * 128 * 128);  // x0 / resize1 / u1
    float*  bufB  = alloc((size_t)8 * 130 * 128 * 128);  // x1 / u0 / y
    float*  bufC  = alloc((size_t)8 * 261 * 64 * 64);    // x2 / t1
    float*  bufD  = alloc((size_t)8 * 261 * 64 * 64);    // xc
    float*  bufE  = alloc((size_t)8 * 261 * 64 * 64);    // t2
    float*  bufF  = alloc((size_t)8 * 130 * 256 * 256);  // resize2
    __bf16* wpack = (__bf16*)alloc(524288);              // 1M bf16 (max 940K needed)
    __bf16* xpack = (__bf16*)alloc(XPACK_CAP / 2);       // 80M bf16

    // ---- encoder ----
    conv_packed(im, conv0_w, nullptr, bufA, B, 3, 256, 256, 64, 7, 7, 1, 3,
                wpack, xpack, stream);
    launch_in(bufA, bufA, in0_g, in0_b, B, 64, 256 * 256, 1, stream);
    conv_packed(bufA, down1_w, nullptr, bufB, B, 64, 256, 256, 128, 4, 4, 2, 1,
                wpack, xpack, stream);
    launch_in(bufB, bufB, in1_g, in1_b, B, 128, 128 * 128, 1, stream);
    conv_packed(bufB, down2_w, nullptr, bufC, B, 128, 128, 128, 256, 4, 4, 2, 1,
                wpack, xpack, stream);
    launch_in(bufC, bufC, in2_g, in2_b, B, 256, 64 * 64, 1, stream);

    // ---- concat + conditional renorm ----
    {
        size_t tot = (size_t)B * CUR * 64 * 64;
        concat_kernel<<<ew_blocks(tot), 256, 0, stream>>>(bufC, c, bufD, 256, CD, 64 * 64, tot);
        meanstd_kernel<<<B * CUR, 256, 0, stream>>>(bufD, mean, stdv, 64 * 64);
        scan_kernel<<<1, 256, 0, stream>>>(mean, stdv, c_org, rmean, rstd, rmf, rsf, B, CD, CUR);
        target_kernel<<<ew_blocks((size_t)B * CUR), 256, 0, stream>>>(c, rmf, rsf, tmean, tstd, B, CD, CUR);
        renorm_kernel<<<ew_blocks(tot), 256, 0, stream>>>(bufD, mean, stdv, tmean, tstd, 64 * 64, tot);
    }

    // ---- 6 residual blocks @ [8,261,64,64] ----
    for (int i = 0; i < 6; ++i) {
        const float* w1 = res_w1 + (size_t)i * CUR * CUR * 9;
        const float* w2 = res_w2 + (size_t)i * CUR * CUR * 9;
        conv_packed(bufD, w1, nullptr, bufC, B, CUR, 64, 64, CUR, 3, 3, 1, 1,
                    wpack, xpack, stream);
        launch_in(bufC, bufC, res_g1 + i * CUR, res_b1 + i * CUR, B, CUR, 64 * 64, 1, stream);
        conv_packed(bufC, w2, nullptr, bufE, B, CUR, 64, 64, CUR, 3, 3, 1, 1,
                    wpack, xpack, stream);
        launch_in(bufE, bufE, res_g2 + i * CUR, res_b2 + i * CUR, B, CUR, 64 * 64, 0, stream);
        size_t tot = (size_t)B * CUR * 64 * 64;
        add_kernel<<<ew_blocks(tot), 256, 0, stream>>>(bufD, bufE, tot);
    }

    // ---- up block 0: resize 64->128, conv 261->130 (+bias), IN, relu ----
    {
        size_t tot = (size_t)B * CUR * 128 * 128;
        resize2x_kernel<<<ew_blocks(tot), 256, 0, stream>>>(bufD, bufA, 64, 64, tot);
        conv_packed(bufA, up0_w, up0_b, bufB, B, CUR, 128, 128, 130, 5, 5, 1, 2,
                    wpack, xpack, stream);
        launch_in(bufB, bufB, up0_g, up0_bt, B, 130, 128 * 128, 1, stream);
    }
    // ---- up block 1: resize 128->256, conv 130->65 (+bias), IN, relu ----
    {
        size_t tot = (size_t)B * 130 * 256 * 256;
        resize2x_kernel<<<ew_blocks(tot), 256, 0, stream>>>(bufB, bufF, 128, 128, tot);
        conv_packed(bufF, up1_w, up1_b, bufA, B, 130, 256, 256, 65, 5, 5, 1, 2,
                    wpack, xpack, stream);
        launch_in(bufA, bufA, up1_g, up1_bt, B, 65, 256 * 256, 1, stream);
    }

    // ---- output conv 65->12 (+bias) into bufB, then polynomial recombination ----
    conv_packed(bufA, out_w, out_b, bufB, B, 65, 256, 256, 12, 7, 7, 1, 3,
                wpack, xpack, stream);
    {
        size_t tot = (size_t)B * 3 * 256 * 256;
        poly_kernel<<<ew_blocks(tot), 256, 0, stream>>>(bufB, im, out, 256 * 256, tot);
    }
}
